// Attention_12214886990302
// MI455X (gfx1250) — matrix operand order repaired and checked by reading, not compiled, not run on hardware
//
#include <hip/hip_runtime.h>

// ---------------------------------------------------------------------------
// Swin window attention forward for MI455X (gfx1250).
// bf16 WMMA everywhere + GLOBAL_LOAD_ASYNC_TO_LDS staging of shared operands.
// ---------------------------------------------------------------------------

typedef __attribute__((ext_vector_type(16))) __bf16 v16bf;
typedef __attribute__((ext_vector_type(8)))  __bf16 v8bf;
typedef __attribute__((ext_vector_type(8)))  float  v8f;

#define NSEQ    196              // 14*14
#define NHEADS  16
#define HDIM    32
#define CIN     512
#define BATCH   128
#define MROWS   (BATCH * NSEQ)   // 25088
#define N_QKV   (3 * NHEADS * HDIM) // 1536
#define BHTOT   (BATCH * NHEADS) // 2048
#define VT_LD   224              // padded vT row length (7 * 32)
#define SCLD    224              // LDS P-strip width (13*16=208, padded to 224)

// LDS layout for k_attn (bytes)
#define ATTN_P_BYTES   (13 * 16 * SCLD * 2)          // 93184
#define ATTN_K_OFF     ATTN_P_BYTES                  // 93184
#define ATTN_K_BYTES   (NSEQ * HDIM * 2)             // 12544
#define ATTN_V_OFF     (ATTN_K_OFF + ATTN_K_BYTES)   // 105728
#define ATTN_V_BYTES   (HDIM * VT_LD * 2)            // 14336
#define ATTN_LDS_BYTES (ATTN_V_OFF + ATTN_V_BYTES)   // 120064

// LDS for GEMM weight strips: 64 cols x 512 K x bf16
#define GEMM_LDS_BYTES (64 * CIN * 2)                // 65536

static __device__ __forceinline__ v8f wmma_bf16(v16bf a, v16bf b, v8f c) {
    // 8 args: (neg_a, A, neg_b, B, c_mod, C, reuse_a, reuse_b)
    return __builtin_amdgcn_wmma_f32_16x16x32_bf16(false, a, false, b,
                                                   (short)0, c, false, false);
}

// Async global -> LDS copy, 16B per lane, tracked by ASYNCcnt.
static __device__ __forceinline__ void async_b128(unsigned lds_off, const void* g) {
    asm volatile("global_load_async_to_lds_b128 %0, %1, off"
                 :: "v"(lds_off), "v"(g) : "memory");
}
static __device__ __forceinline__ void wait_async0() {
    asm volatile("s_wait_asynccnt 0" ::: "memory");
}

// A fragment (16x32 bf16, row-major source):
// lane = l15 -> row; K runs {hi*16..hi*16+15} (contiguous, matches B loads).
template <typename PTR>
static __device__ __forceinline__ v16bf load_afrag(PTR rowp, int kb, int hi) {
    v8bf lo = *(const v8bf*)(rowp + kb + hi * 16);
    v8bf hv = *(const v8bf*)(rowp + kb + hi * 16 + 8);
    v16bf a;
#pragma unroll
    for (int i = 0; i < 8; ++i) { a[i] = lo[i]; a[8 + i] = hv[i]; }
    return a;
}

// ------------------------- prepass kernels ---------------------------------

__global__ void k_cvt(const float* __restrict__ in, __bf16* __restrict__ out, int n) {
    int i = blockIdx.x * blockDim.x + threadIdx.x;
    if (i < n) out[i] = (__bf16)in[i];
}

// in[rows][cols] fp32 -> out[cols][rows] bf16  (weights transposed so the
// WMMA B-fragment is a contiguous 32B load per lane)
__global__ void k_cvt_t(const float* __restrict__ in, __bf16* __restrict__ out,
                        int rows, int cols) {
    int i = blockIdx.x * blockDim.x + threadIdx.x;
    if (i < rows * cols) {
        int r = i / cols, c = i - r * cols;
        out[(size_t)c * rows + r] = (__bf16)in[i];
    }
}

// bias_pre[h][m][c] = rel_bias_table[rel_pos_index[m][c]][h]
__global__ void k_bias(const float* __restrict__ tbl, const int* __restrict__ rpi,
                       float* __restrict__ bp) {
    int i = blockIdx.x * blockDim.x + threadIdx.x;
    if (i < NHEADS * NSEQ * NSEQ) {
        int h   = i / (NSEQ * NSEQ);
        int rem = i - h * (NSEQ * NSEQ);
        bp[i] = tbl[rpi[rem] * NHEADS + h];
    }
}

// ------------------------- QKV projection GEMM -----------------------------
// [25088 x 512] x [512 x 1536]; block = 8 waves; each wave owns TWO 16-row
// m-tiles over the same 64-column strip (B-fragments reused across both);
// strip weights async-staged to LDS once per block.

__global__ void k_gemm_qkv(const __bf16* __restrict__ xb,
                           const __bf16* __restrict__ wT,   // [1536][512]
                           __bf16* __restrict__ q,
                           __bf16* __restrict__ kmat,
                           __bf16* __restrict__ vT) {
    extern __shared__ __bf16 Wl[];                   // [64][512]
    const int lane = threadIdx.x & 31;
    const int wave = threadIdx.x >> 5;      // 0..7
    const int l15  = lane & 15;
    const int hi   = (lane >> 4) & 1;
    const int mt0  = (blockIdx.x * 16 + wave * 2) * 16;   // first m-tile
    const int n0   = blockIdx.y * 64;

    // async-stage the 64x512 bf16 weight strip (contiguous 64KB of wT)
    {
        const unsigned wbase = (unsigned)(size_t)Wl;
        const char* wsrc = (const char*)(wT + (size_t)n0 * CIN);
        for (int i = threadIdx.x; i < GEMM_LDS_BYTES / 16; i += blockDim.x)
            async_b128(wbase + i * 16, wsrc + i * 16);
        wait_async0();
        __syncthreads();
    }

    const __bf16* ap0 = xb + (size_t)(mt0 + l15) * CIN;
    const __bf16* ap1 = ap0 + (size_t)16 * CIN;

    float acc[2][4][8];
#pragma unroll
    for (int mi = 0; mi < 2; ++mi)
#pragma unroll
        for (int t = 0; t < 4; ++t)
#pragma unroll
            for (int r = 0; r < 8; ++r) acc[mi][t][r] = 0.f;

    for (int kb = 0; kb < CIN; kb += 32) {
        // all 4 B fragments first (partial dscnt waits instead of full drain)
        v16bf bf[4];
#pragma unroll
        for (int t = 0; t < 4; ++t)
            bf[t] = *(const v16bf*)(Wl + (t * 16 + l15) * CIN + kb + hi * 16);
        v16bf af0 = load_afrag(ap0, kb, hi);
        v16bf af1 = load_afrag(ap1, kb, hi);
#pragma unroll
        for (int t = 0; t < 4; ++t) {
            v8f c;
#pragma unroll
            for (int r = 0; r < 8; ++r) c[r] = acc[0][t][r];
            c = wmma_bf16(af0, bf[t], c);
#pragma unroll
            for (int r = 0; r < 8; ++r) acc[0][t][r] = c[r];
        }
#pragma unroll
        for (int t = 0; t < 4; ++t) {
            v8f c;
#pragma unroll
            for (int r = 0; r < 8; ++r) c[r] = acc[1][t][r];
            c = wmma_bf16(af1, bf[t], c);
#pragma unroll
            for (int r = 0; r < 8; ++r) acc[1][t][r] = c[r];
        }
    }

    const float qscale = 0.17677669529663687f;   // (512/16)^-0.5 = 32^-0.5
#pragma unroll
    for (int mi = 0; mi < 2; ++mi) {
        const int m0 = mt0 + mi * 16;
#pragma unroll
        for (int t = 0; t < 4; ++t) {
            int col = n0 + t * 16 + l15;
            int s   = col >> 9;        // 0=q, 1=k, 2=v
            int rem = col & 511;
            int h   = rem >> 5;
            int d   = rem & 31;
#pragma unroll
            for (int r = 0; r < 8; ++r) {
                int row = m0 + r + hi * 8;       // global token row
                int b   = row / NSEQ;
                int ni  = row - b * NSEQ;
                int bh  = b * NHEADS + h;
                float val = acc[mi][t][r];
                if (s == 0)
                    q[((size_t)bh * NSEQ + ni) * HDIM + d] = (__bf16)(val * qscale);
                else if (s == 1)
                    kmat[((size_t)bh * NSEQ + ni) * HDIM + d] = (__bf16)val;
                else
                    vT[((size_t)bh * HDIM + d) * VT_LD + ni] = (__bf16)val;
            }
        }
    }
}

// ------------------------- fused attention ---------------------------------
// One workgroup per (b,h); 13 waves, each owns a 16-row query strip.
// K and V^T async-staged to LDS once per workgroup (shared by all 13 waves).
// QK^T: 13 WMMAs (K=32 = one k-step). Softmax exact (full row in VGPRs).
// P -> wave-private LDS strip (bf16) -> A-fragments for AV (7 k-steps x 2).

__global__ void k_attn(const __bf16* __restrict__ q,
                       const __bf16* __restrict__ kmat,
                       const __bf16* __restrict__ vT,
                       const float*  __restrict__ biasp,   // [16][196][196]
                       __bf16* __restrict__ ob) {           // [25088][512]
    extern __shared__ __bf16 Pl[];   // [13][16][224] P-strips, then K, then V^T
    const int lane = threadIdx.x & 31;
    const int wave = threadIdx.x >> 5;      // 0..12
    const int l15  = lane & 15;
    const int hi   = (lane >> 4) & 1;
    const int bh   = blockIdx.x;
    const int b    = bh >> 4;
    const int h    = bh & 15;
    const int m0   = wave * 16;

    const __bf16* qp = q    + (size_t)bh * NSEQ * HDIM;
    const __bf16* kp = kmat + (size_t)bh * NSEQ * HDIM;
    const __bf16* vp = vT   + (size_t)bh * HDIM * VT_LD;
    __bf16* strip = Pl + wave * 16 * SCLD;
    const __bf16* kls = Pl + ATTN_K_OFF / 2;   // [196][32]
    const __bf16* vls = Pl + ATTN_V_OFF / 2;   // [32][224]

    // async-stage K (12544B) and V^T (14336B) into LDS, all waves cooperate
    {
        const unsigned base = (unsigned)(size_t)Pl;
        for (int i = threadIdx.x; i < ATTN_K_BYTES / 16; i += blockDim.x)
            async_b128(base + ATTN_K_OFF + i * 16, (const char*)kp + i * 16);
        for (int i = threadIdx.x; i < ATTN_V_BYTES / 16; i += blockDim.x)
            async_b128(base + ATTN_V_OFF + i * 16, (const char*)vp + i * 16);
        wait_async0();
        __syncthreads();
    }

    // Q fragment for this strip (rows clamped; padded rows masked later)
    int qrow = m0 + l15; if (qrow > NSEQ - 1) qrow = NSEQ - 1;
    v16bf af = load_afrag(qp + qrow * HDIM, 0, hi);

    float sc[13][8];
#pragma unroll
    for (int t = 0; t < 13; ++t) {
        int kc = t * 16 + l15; if (kc > NSEQ - 1) kc = NSEQ - 1;
        v16bf bf = *(const v16bf*)(kls + kc * HDIM + hi * 16);
        v8f c;
#pragma unroll
        for (int r = 0; r < 8; ++r) c[r] = 0.f;
        c = wmma_bf16(af, bf, c);
#pragma unroll
        for (int r = 0; r < 8; ++r) sc[t][r] = c[r];
    }

    // + relative-position bias, mask out-of-range keys/queries
    const float* bp = biasp + (size_t)h * NSEQ * NSEQ;
#pragma unroll
    for (int t = 0; t < 13; ++t) {
        int col = t * 16 + l15;
#pragma unroll
        for (int r = 0; r < 8; ++r) {
            int m = m0 + r + hi * 8;
            if (m < NSEQ && col < NSEQ) sc[t][r] += bp[m * NSEQ + col];
            else                        sc[t][r]  = -1e30f;
        }
    }

    // exact softmax: reduce over 13 tiles + 16 lanes of this half-wave
#pragma unroll
    for (int r = 0; r < 8; ++r) {
        float mx = -1e30f;
#pragma unroll
        for (int t = 0; t < 13; ++t) mx = fmaxf(mx, sc[t][r]);
        mx = fmaxf(mx, __shfl_xor(mx, 1));
        mx = fmaxf(mx, __shfl_xor(mx, 2));
        mx = fmaxf(mx, __shfl_xor(mx, 4));
        mx = fmaxf(mx, __shfl_xor(mx, 8));
        float sum = 0.f;
#pragma unroll
        for (int t = 0; t < 13; ++t) {
            float e = __expf(sc[t][r] - mx);
            sc[t][r] = e;
            sum += e;
        }
        sum += __shfl_xor(sum, 1);
        sum += __shfl_xor(sum, 2);
        sum += __shfl_xor(sum, 4);
        sum += __shfl_xor(sum, 8);
        float inv = 1.f / sum;
#pragma unroll
        for (int t = 0; t < 13; ++t) sc[t][r] *= inv;
    }

    // P -> wave-private LDS strip (C-layout scatter; no barrier needed)
#pragma unroll
    for (int t = 0; t < 13; ++t) {
        int col = t * 16 + l15;
#pragma unroll
        for (int r = 0; r < 8; ++r)
            strip[(r + hi * 8) * SCLD + col] = (__bf16)sc[t][r];
    }
    {   // zero the pad columns 208..223 so the last AV k-step contributes 0
        int col = 208 + l15;
#pragma unroll
        for (int r = 0; r < 8; ++r)
            strip[(r + hi * 8) * SCLD + col] = (__bf16)0.f;
    }

    // AV: out[16 x 32] = P[16 x 224] x V[224 x 32], 7 k-steps x 2 n-tiles
#pragma unroll
    for (int nt = 0; nt < 2; ++nt) {
        v8f c;
#pragma unroll
        for (int r = 0; r < 8; ++r) c[r] = 0.f;
        for (int kkb = 0; kkb < VT_LD; kkb += 32) {
            v16bf pa = load_afrag(strip + l15 * SCLD, kkb, hi);
            v16bf vb = *(const v16bf*)(vls + (nt * 16 + l15) * VT_LD
                                           + kkb + hi * 16);
            c = wmma_bf16(pa, vb, c);
        }
#pragma unroll
        for (int r = 0; r < 8; ++r) {
            int m = m0 + r + hi * 8;
            if (m < NSEQ) {
                size_t row = (size_t)b * NSEQ + m;
                ob[row * (NHEADS * HDIM) + h * HDIM + nt * 16 + l15] = (__bf16)c[r];
            }
        }
    }
}

// ------------------------- output projection GEMM --------------------------
// [25088 x 512] x [512 x 512] + bias, fp32 result. Same structure as QKV GEMM.

__global__ void k_gemm_proj(const __bf16* __restrict__ ob,
                            const __bf16* __restrict__ wT,   // [512][512]
                            const float*  __restrict__ bias,
                            float* __restrict__ out) {
    extern __shared__ __bf16 Wl[];                   // [64][512]
    const int lane = threadIdx.x & 31;
    const int wave = threadIdx.x >> 5;      // 0..7
    const int l15  = lane & 15;
    const int hi   = (lane >> 4) & 1;
    const int mt0  = (blockIdx.x * 16 + wave * 2) * 16;
    const int n0   = blockIdx.y * 64;

    {
        const unsigned wbase = (unsigned)(size_t)Wl;
        const char* wsrc = (const char*)(wT + (size_t)n0 * CIN);
        for (int i = threadIdx.x; i < GEMM_LDS_BYTES / 16; i += blockDim.x)
            async_b128(wbase + i * 16, wsrc + i * 16);
        wait_async0();
        __syncthreads();
    }

    const __bf16* ap0 = ob + (size_t)(mt0 + l15) * CIN;
    const __bf16* ap1 = ap0 + (size_t)16 * CIN;

    float acc[2][4][8];
#pragma unroll
    for (int mi = 0; mi < 2; ++mi)
#pragma unroll
        for (int t = 0; t < 4; ++t)
#pragma unroll
            for (int r = 0; r < 8; ++r) acc[mi][t][r] = 0.f;

    for (int kb = 0; kb < CIN; kb += 32) {
        v16bf bf[4];
#pragma unroll
        for (int t = 0; t < 4; ++t)
            bf[t] = *(const v16bf*)(Wl + (t * 16 + l15) * CIN + kb + hi * 16);
        v16bf af0 = load_afrag(ap0, kb, hi);
        v16bf af1 = load_afrag(ap1, kb, hi);
#pragma unroll
        for (int t = 0; t < 4; ++t) {
            v8f c;
#pragma unroll
            for (int r = 0; r < 8; ++r) c[r] = acc[0][t][r];
            c = wmma_bf16(af0, bf[t], c);
#pragma unroll
            for (int r = 0; r < 8; ++r) acc[0][t][r] = c[r];
        }
#pragma unroll
        for (int t = 0; t < 4; ++t) {
            v8f c;
#pragma unroll
            for (int r = 0; r < 8; ++r) c[r] = acc[1][t][r];
            c = wmma_bf16(af1, bf[t], c);
#pragma unroll
            for (int r = 0; r < 8; ++r) acc[1][t][r] = c[r];
        }
    }

#pragma unroll
    for (int mi = 0; mi < 2; ++mi) {
        const int m0 = mt0 + mi * 16;
#pragma unroll
        for (int t = 0; t < 4; ++t) {
            int col = n0 + t * 16 + l15;
#pragma unroll
            for (int r = 0; r < 8; ++r) {
                int row = m0 + r + hi * 8;
                out[(size_t)row * CIN + col] = acc[mi][t][r] + bias[col];
            }
        }
    }
}

// ------------------------- launcher ----------------------------------------

extern "C" void kernel_launch(void* const* d_in, const int* in_sizes, int n_in,
                              void* d_out, int out_size, void* d_ws, size_t ws_size,
                              hipStream_t stream) {
    (void)in_sizes; (void)n_in; (void)out_size; (void)ws_size;

    const float* x      = (const float*)d_in[0];   // [128,196,512]
    const float* qkv_w  = (const float*)d_in[1];   // [512,1536]
    const float* tbl    = (const float*)d_in[2];   // [729,16]
    const float* proj_w = (const float*)d_in[3];   // [512,512]
    const float* proj_b = (const float*)d_in[4];   // [512]
    const int*   rpi    = (const int*)d_in[5];     // [196,196]
    float* out = (float*)d_out;

    // workspace carve-up (~137 MB total)
    char* ws = (char*)d_ws;
    size_t off = 0;
    auto carve = [&](size_t bytes) -> char* {
        char* p = ws + off;
        off += (bytes + 255) & ~(size_t)255;
        return p;
    };
    __bf16* xb     = (__bf16*)carve((size_t)MROWS * CIN * 2);       // 25.7 MB
    __bf16* qkvwT  = (__bf16*)carve((size_t)N_QKV * CIN * 2);       // 1.6 MB
    __bf16* projwT = (__bf16*)carve((size_t)CIN * CIN * 2);         // 0.5 MB
    __bf16* qb     = (__bf16*)carve((size_t)BHTOT * NSEQ * HDIM * 2);
    __bf16* kb     = (__bf16*)carve((size_t)BHTOT * NSEQ * HDIM * 2);
    __bf16* vTb    = (__bf16*)carve((size_t)BHTOT * HDIM * VT_LD * 2);
    __bf16* obuf   = (__bf16*)carve((size_t)MROWS * CIN * 2);
    float*  biasp  = (float*)carve((size_t)NHEADS * NSEQ * NSEQ * 4);

    int nx = MROWS * CIN;
    k_cvt  <<<(nx + 255) / 256, 256, 0, stream>>>(x, xb, nx);
    k_cvt_t<<<(CIN * N_QKV + 255) / 256, 256, 0, stream>>>(qkv_w, qkvwT, CIN, N_QKV);
    k_cvt_t<<<(CIN * CIN + 255) / 256, 256, 0, stream>>>(proj_w, projwT, CIN, CIN);
    k_bias <<<(NHEADS * NSEQ * NSEQ + 255) / 256, 256, 0, stream>>>(tbl, rpi, biasp);

    // 256 output rows per block (8 waves x 2 m-tiles x 16 rows)
    k_gemm_qkv<<<dim3(MROWS / 256, N_QKV / 64), 256, GEMM_LDS_BYTES, stream>>>
        (xb, qkvwT, qb, kb, vTb);

    k_attn<<<BHTOT, 13 * 32, ATTN_LDS_BYTES, stream>>>(qb, kb, vTb, biasp, obuf);

    k_gemm_proj<<<dim3(MROWS / 256, CIN / 64), 256, GEMM_LDS_BYTES, stream>>>
        (obuf, projwT, proj_b, out);
}
